// TimeSformer_50878182588514
// MI455X (gfx1250) — matrix operand order repaired and checked by reading, not compiled, not run on hardware
//
#include <hip/hip_runtime.h>
#include <hip/hip_bf16.h>

// ---------------------------------------------------------------------------
// TimeSformer forward for MI455X (gfx1250). All GEMMs run through a generic
// NT WMMA kernel (v_wmma_f32_16x16x32_f16): A[M,K] @ W[N,K]^T + bias, f32 in,
// f16 staged via LDS, f32 accumulate. Attention matmuls reuse the same kernel
// by storing v transposed (vT[hd, L]).
// Staging: interior tiles use unguarded global_load_b128 + packed f16 convert
// + ds_store_b128; edge tiles use clamped addresses + cndmask selects.
// Next K-tile is prefetched (global_prefetch_b8) while WMMAs run.
// ---------------------------------------------------------------------------

typedef __attribute__((ext_vector_type(16))) _Float16 v16h;
typedef __attribute__((ext_vector_type(8)))  _Float16 h8;
typedef __attribute__((ext_vector_type(8)))  float    v8f;
typedef __attribute__((ext_vector_type(4)))  float    f4;

enum {
  BB = 2, TT = 8, CC = 768, HHD = 12, HD = 64, NP = 196,
  SEQ = 1569,           // 1 + T*N
  BTN = 3136,           // B*T*N
  RS  = 3152,           // B*T*197
  LT  = 8, LS = 197,
  BHT = 4704,           // B*N*heads
  BHS = 192,            // B*T*heads
  ROWS = 3138           // B*SEQ
};

#define TILE 64
#define KSTEP 32
#define LDH 40   // LDS row stride in halves (32 + 8 pad, keeps 16B alignment)

__device__ __forceinline__ float gelu_exact(float x) {
  return 0.5f * x * (1.0f + erff(x * 0.70710678118654752f));
}

// Stage one 8-float chunk (row `gr` of a [rows x K] f32 matrix, cols
// k0+lc .. k0+lc+7) into LDS as f16.  `full` and `aligned` are
// block-uniform so the branch is scalar.
__device__ __forceinline__ void stage8(const float* __restrict__ src, int rows, int K,
                                       int gr, int k0, int lc, _Float16* __restrict__ dst,
                                       bool full, bool aligned)
{
  float va[8];
  if (full & aligned) {
    const f4* p4 = (const f4*)(src + (size_t)gr * K + (k0 + lc));
    f4 u0 = p4[0];
    f4 u1 = p4[1];
#pragma unroll
    for (int e = 0; e < 4; ++e) { va[e] = u0[e]; va[e + 4] = u1[e]; }
  } else if (full) {
    const float* p = src + (size_t)gr * K + (k0 + lc);
#pragma unroll
    for (int e = 0; e < 8; ++e) va[e] = p[e];
  } else {
    // Clamped addressing + select: loads always legal, no exec divergence.
    int grc = gr < rows ? gr : (rows - 1);
    const float* p = src + (size_t)grc * K;
#pragma unroll
    for (int e = 0; e < 8; ++e) {
      int kk = k0 + lc + e;
      int kkc = kk < K ? kk : (K - 1);
      float v = p[kkc];
      va[e] = (gr < rows && kk < K) ? v : 0.f;
    }
  }
  h8 hv;
#pragma unroll
  for (int e = 0; e < 8; ++e) hv[e] = (_Float16)va[e];
  *(h8*)dst = hv;   // single ds_store_b128
}

// ------------------------------ WMMA GEMM ----------------------------------
template<int EPI>  // 0 = none, 1 = exact gelu
__global__ __launch_bounds__(256)
void gemm_nt(const float* __restrict__ A, const float* __restrict__ Bw,
             const float* __restrict__ bias, float* __restrict__ Cout,
             int M, int N, int K,
             long long sA, long long sB, long long sC)
{
  __shared__ _Float16 Al[TILE * LDH];
  __shared__ _Float16 Bl[TILE * LDH];

  const int z = blockIdx.z;
  A    += (size_t)z * sA;
  Bw   += (size_t)z * sB;
  Cout += (size_t)z * sC;

  const int tM = blockIdx.y * TILE;
  const int tN = blockIdx.x * TILE;
  const int tid  = threadIdx.x;
  const int lane = tid & 31;
  const int wave = tid >> 5;
  const int m0 = (wave & 3) * 16;        // subtile row in tile
  const int n0 = (wave >> 2) * 32;       // subtile col base (2 subtiles/wave)

  v8f acc0 = {0.f,0.f,0.f,0.f,0.f,0.f,0.f,0.f};
  v8f acc1 = {0.f,0.f,0.f,0.f,0.f,0.f,0.f,0.f};

  const int lr = tid >> 2;          // staging row 0..63
  const int lc = (tid & 3) * 8;     // staging col base 0..24
  const int lh = lane & 15;
  const int hi = lane >> 4;         // lane-half select

  const bool aligned  = ((K & 3) == 0);
  const bool aRowFull = (tM + TILE <= M);
  const bool bRowFull = (tN + TILE <= N);

  // Clamped rows for staging/prefetch address legality.
  const int arc = (tM + lr) < M ? (tM + lr) : (M - 1);
  const int brc = (tN + lr) < N ? (tN + lr) : (N - 1);

  for (int k0 = 0; k0 < K; k0 += KSTEP) {
    const bool kFull = (k0 + KSTEP <= K);
    __syncthreads();
    stage8(A,  M, K, tM + lr, k0, lc, &Al[lr * LDH + lc], aRowFull && kFull, aligned);
    stage8(Bw, N, K, tN + lr, k0, lc, &Bl[lr * LDH + lc], bRowFull && kFull, aligned);

    // Prefetch next K-tile into cache while this tile's WMMAs execute
    // (global_prefetch_b8: speculative, no counter, whole-cacheline fetch).
    if (k0 + KSTEP < K) {
      int kp = k0 + KSTEP + lc;
      if (kp < K) {
        __builtin_prefetch(A  + (size_t)arc * K + kp, 0, 3);
        __builtin_prefetch(Bw + (size_t)brc * K + kp, 0, 3);
      }
    }
    __syncthreads();

    // ---- A fragment: lanes 0-15 -> K 0..15, lanes 16-31 -> K 16..31 ----
    const _Float16* ap = &Al[(m0 + lh) * LDH + hi * 16];
    h8 alo = *(const h8*)(ap);
    h8 ahi = *(const h8*)(ap + 8);
    v16h af;
#pragma unroll
    for (int i = 0; i < 8; ++i) { af[i] = alo[i]; af[i + 8] = ahi[i]; }

    // ---- B fragments: lane n = lane&15, K range 0-15 / 16-31 by lane-half --
    const _Float16* bp0 = &Bl[(n0 + lh) * LDH + hi * 16];
    h8 b0lo = *(const h8*)(bp0);
    h8 b0hi = *(const h8*)(bp0 + 8);
    v16h bf0;
#pragma unroll
    for (int i = 0; i < 8; ++i) { bf0[i] = b0lo[i]; bf0[i + 8] = b0hi[i]; }

    const _Float16* bp1 = &Bl[(n0 + 16 + lh) * LDH + hi * 16];
    h8 b1lo = *(const h8*)(bp1);
    h8 b1hi = *(const h8*)(bp1 + 8);
    v16h bf1;
#pragma unroll
    for (int i = 0; i < 8; ++i) { bf1[i] = b1lo[i]; bf1[i + 8] = b1hi[i]; }

    acc0 = __builtin_amdgcn_wmma_f32_16x16x32_f16(false, af, false, bf0,
                                                  (short)0, acc0, false, false);
    acc1 = __builtin_amdgcn_wmma_f32_16x16x32_f16(false, af, false, bf1,
                                                  (short)0, acc1, false, false);
  }

  // ---- epilogue: C/D layout row = hi*8 + r, col = lane&15 ----
#pragma unroll
  for (int r = 0; r < 8; ++r) {
    int gr  = tM + m0 + hi * 8 + r;
    int gc0 = tN + n0 + lh;
    int gc1 = gc0 + 16;
    if (gr < M) {
      if (gc0 < N) {
        float v = acc0[r] + (bias ? bias[gc0] : 0.f);
        if (EPI == 1) v = gelu_exact(v);
        Cout[(size_t)gr * N + gc0] = v;
      }
      if (gc1 < N) {
        float v = acc1[r] + (bias ? bias[gc1] : 0.f);
        if (EPI == 1) v = gelu_exact(v);
        Cout[(size_t)gr * N + gc1] = v;
      }
    }
  }
}

// ------------------------------ LayerNorm ----------------------------------
__global__ __launch_bounds__(256)
void ln_kernel(const float* __restrict__ in, const float* __restrict__ w,
               const float* __restrict__ b, float* __restrict__ out, int C)
{
  const size_t row = blockIdx.x;
  const float* xr = in + row * C;
  float* o = out + row * C;
  __shared__ float red[256];
  const int tid = threadIdx.x;

  float s = 0.f;
  for (int i = tid; i < C; i += 256) s += xr[i];
  red[tid] = s; __syncthreads();
  for (int st = 128; st > 0; st >>= 1) {
    if (tid < st) red[tid] += red[tid + st];
    __syncthreads();
  }
  float mean = red[0] / C;
  __syncthreads();

  float v = 0.f;
  for (int i = tid; i < C; i += 256) { float d = xr[i] - mean; v += d * d; }
  red[tid] = v; __syncthreads();
  for (int st = 128; st > 0; st >>= 1) {
    if (tid < st) red[tid] += red[tid + st];
    __syncthreads();
  }
  float rstd = rsqrtf(red[0] / C + 1e-6f);

  for (int i = tid; i < C; i += 256)
    o[i] = (xr[i] - mean) * rstd * w[i] + b[i];
}

// ------------------------------ Softmax ------------------------------------
__global__ __launch_bounds__(256)
void softmax_kernel(float* __restrict__ s, int L, float scale)
{
  float* p = s + (size_t)blockIdx.x * L;
  __shared__ float red[256];
  const int tid = threadIdx.x;

  float m = -3.4e38f;
  for (int i = tid; i < L; i += 256) m = fmaxf(m, p[i] * scale);
  red[tid] = m; __syncthreads();
  for (int st = 128; st > 0; st >>= 1) {
    if (tid < st) red[tid] = fmaxf(red[tid], red[tid + st]);
    __syncthreads();
  }
  float mx = red[0];
  __syncthreads();

  float sum = 0.f;
  for (int i = tid; i < L; i += 256) {
    float e = expf(p[i] * scale - mx);
    p[i] = e; sum += e;
  }
  red[tid] = sum; __syncthreads();
  for (int st = 128; st > 0; st >>= 1) {
    if (tid < st) red[tid] += red[tid + st];
    __syncthreads();
  }
  float inv = 1.f / red[0];
  for (int i = tid; i < L; i += 256) p[i] *= inv;
}

// --------------------------- Elementwise kernels ---------------------------
__global__ void im2row_kernel(const float* __restrict__ x, float* __restrict__ out, int total)
{
  int idx = blockIdx.x * 256 + threadIdx.x;
  if (idx >= total) return;
  int k = idx % CC, row = idx / CC;
  int n = row % NP, bt = row / NP;
  int t = bt % TT, b = bt / TT;
  int ci = k >> 8, rem = k & 255, py = rem >> 4, px = rem & 15;
  int nh = n / 14, nw = n % 14;
  size_t src = ((((size_t)b * 3 + ci) * TT + t) * 224 + (nh * 16 + py)) * 224 + (nw * 16 + px);
  out[idx] = x[src];
}

__global__ void assemble_kernel(const float* __restrict__ xe, const float* __restrict__ cls_token,
                                const float* __restrict__ pos, const float* __restrict__ timeE,
                                float* __restrict__ X, int total)
{
  int idx = blockIdx.x * 256 + threadIdx.x;
  if (idx >= total) return;
  int c = idx % CC, row = idx / CC;
  int b = row / SEQ, p = row % SEQ;
  float v;
  if (p == 0) {
    v = cls_token[c] + pos[c];
  } else {
    int i = p - 1, t = i / NP, n = i % NP;
    v = xe[((size_t)(b * TT + t) * NP + n) * CC + c]
      + pos[(size_t)(1 + n) * CC + c] + timeE[(size_t)t * CC + c];
  }
  X[idx] = v;
}

__global__ void split_t_kernel(const float* __restrict__ qkv, float* __restrict__ q,
                               float* __restrict__ k, float* __restrict__ vt, int total)
{
  int idx = blockIdx.x * 256 + threadIdx.x;
  if (idx >= total) return;
  int hc = idx % CC, row = idx / CC;
  int h = hc >> 6, dd = hc & 63;
  int i = row % (TT * NP), b = row / (TT * NP);
  int t = i / NP, n = i % NP;
  int bh = (b * NP + n) * HHD + h;
  const float* src = qkv + (size_t)row * (3 * CC) + hc;
  q[((size_t)bh * LT + t) * HD + dd]  = src[0];
  k[((size_t)bh * LT + t) * HD + dd]  = src[CC];
  vt[((size_t)bh * HD + dd) * LT + t] = src[2 * CC];
}

__global__ void split_s_kernel(const float* __restrict__ qkv, float* __restrict__ q,
                               float* __restrict__ k, float* __restrict__ vt, int total)
{
  int idx = blockIdx.x * 256 + threadIdx.x;
  if (idx >= total) return;
  int hc = idx % CC, row = idx / CC;
  int h = hc >> 6, dd = hc & 63;
  int p = row % LS, bt = row / LS;
  int bh = bt * HHD + h;
  const float* src = qkv + (size_t)row * (3 * CC) + hc;
  q[((size_t)bh * LS + p) * HD + dd]  = src[0];
  k[((size_t)bh * LS + p) * HD + dd]  = src[CC];
  vt[((size_t)bh * HD + dd) * LS + p] = src[2 * CC];
}

__global__ void merge_t_kernel(const float* __restrict__ ao, float* __restrict__ mg, int total)
{
  int idx = blockIdx.x * 256 + threadIdx.x;
  if (idx >= total) return;
  int hc = idx % CC, row = idx / CC;
  int h = hc >> 6, dd = hc & 63;
  int i = row % (TT * NP), b = row / (TT * NP);
  int t = i / NP, n = i % NP;
  int bh = (b * NP + n) * HHD + h;
  mg[idx] = ao[((size_t)bh * LT + t) * HD + dd];
}

__global__ void merge_s_kernel(const float* __restrict__ ao, float* __restrict__ mg, int total)
{
  int idx = blockIdx.x * 256 + threadIdx.x;
  if (idx >= total) return;
  int hc = idx % CC, row = idx / CC;
  int h = hc >> 6, dd = hc & 63;
  int p = row % LS, bt = row / LS;
  int bh = bt * HHD + h;
  mg[idx] = ao[((size_t)bh * LS + p) * HD + dd];
}

__global__ void add_body_kernel(float* __restrict__ X, const float* __restrict__ src, int total)
{
  int idx = blockIdx.x * 256 + threadIdx.x;
  if (idx >= total) return;
  int c = idx % CC, row = idx / CC;
  int i = row % (TT * NP), b = row / (TT * NP);
  X[((size_t)b * SEQ + 1 + i) * CC + c] += src[idx];
}

__global__ void build_xs_kernel(const float* __restrict__ X, float* __restrict__ xs, int total)
{
  int idx = blockIdx.x * 256 + threadIdx.x;
  if (idx >= total) return;
  int c = idx % CC, row = idx / CC;
  int p = row % LS, bt = row / LS;
  int b = bt / TT, t = bt % TT;
  float v;
  if (p == 0) v = X[(size_t)b * SEQ * CC + c];
  else {
    int i = t * NP + (p - 1);
    v = X[((size_t)b * SEQ + 1 + i) * CC + c];
  }
  xs[idx] = v;
}

__global__ void combine_body_kernel(float* __restrict__ X, const float* __restrict__ sout, int total)
{
  int idx = blockIdx.x * 256 + threadIdx.x;
  if (idx >= total) return;
  int c = idx % CC, row = idx / CC;
  int i = row % (TT * NP), b = row / (TT * NP);
  int t = i / NP, n = i % NP;
  X[((size_t)b * SEQ + 1 + i) * CC + c] +=
      sout[(((size_t)(b * TT + t)) * LS + 1 + n) * CC + c];
}

__global__ void combine_cls_kernel(float* __restrict__ X, const float* __restrict__ sout, int total)
{
  int idx = blockIdx.x * 256 + threadIdx.x;
  if (idx >= total) return;
  int c = idx % CC, b = idx / CC;
  float s = 0.f;
  for (int t = 0; t < TT; ++t)
    s += sout[((size_t)(b * TT + t) * LS) * CC + c];
  X[(size_t)b * SEQ * CC + c] += s * (1.f / TT);
}

__global__ void add_all_kernel(float* __restrict__ X, const float* __restrict__ src, int total)
{
  int idx = blockIdx.x * 256 + threadIdx.x;
  if (idx >= total) return;
  X[idx] += src[idx];
}

__global__ void gather_cls_kernel(const float* __restrict__ X, float* __restrict__ out, int total)
{
  int idx = blockIdx.x * 256 + threadIdx.x;
  if (idx >= total) return;
  int c = idx % CC, b = idx / CC;
  out[idx] = X[(size_t)b * SEQ * CC + c];
}

// ------------------------------- Host side ---------------------------------
static void run_gemm(hipStream_t stream, const float* A, const float* Bw,
                     const float* bias, float* C, int M, int N, int K,
                     int batches, long long sA, long long sB, long long sC, int epi)
{
  dim3 grid((N + TILE - 1) / TILE, (M + TILE - 1) / TILE, batches);
  if (epi) gemm_nt<1><<<grid, 256, 0, stream>>>(A, Bw, bias, C, M, N, K, sA, sB, sC);
  else     gemm_nt<0><<<grid, 256, 0, stream>>>(A, Bw, bias, C, M, N, K, sA, sB, sC);
}

#define EL(total) (((total) + 255) / 256), 256, 0, stream

extern "C" void kernel_launch(void* const* d_in, const int* in_sizes, int n_in,
                              void* d_out, int out_size, void* d_ws, size_t ws_size,
                              hipStream_t stream)
{
  const float* x          = (const float*)d_in[0];
  const float* conv_w     = (const float*)d_in[1];
  const float* conv_b     = (const float*)d_in[2];
  const float* cls_token  = (const float*)d_in[3];
  const float* pos_embed  = (const float*)d_in[4];
  const float* time_embed = (const float*)d_in[5];
  const float* tn_w   = (const float*)d_in[6];
  const float* tn_b   = (const float*)d_in[7];
  const float* tqkv_w = (const float*)d_in[8];
  const float* tqkv_b = (const float*)d_in[9];
  const float* tproj_w= (const float*)d_in[10];
  const float* tproj_b= (const float*)d_in[11];
  const float* tfc_w  = (const float*)d_in[12];
  const float* tfc_b  = (const float*)d_in[13];
  const float* n1_w   = (const float*)d_in[14];
  const float* n1_b   = (const float*)d_in[15];
  const float* sqkv_w = (const float*)d_in[16];
  const float* sqkv_b = (const float*)d_in[17];
  const float* sproj_w= (const float*)d_in[18];
  const float* sproj_b= (const float*)d_in[19];
  const float* n2_w   = (const float*)d_in[20];
  const float* n2_b   = (const float*)d_in[21];
  const float* fc1_w  = (const float*)d_in[22];
  const float* fc1_b  = (const float*)d_in[23];
  const float* fc2_w  = (const float*)d_in[24];
  const float* fc2_b  = (const float*)d_in[25];
  const float* norm_w = (const float*)d_in[26];
  const float* norm_b = (const float*)d_in[27];
  const float* head_w = (const float*)d_in[28];
  const float* head_b = (const float*)d_in[29];
  (void)in_sizes; (void)n_in; (void)out_size; (void)ws_size;

  float* ws  = (float*)d_ws;
  float* X    = ws;                      // 2*1569*768          = 2,409,984
  float* LNB  = X    + 2409984;          // 3152*768            = 2,420,736
  float* QKV  = LNB  + 2420736;          // 3152*2304           = 7,262,208
  float* Qb   = QKV  + 7262208;          // 2,420,736
  float* Kb   = Qb   + 2420736;          // 2,420,736
  float* VT   = Kb   + 2420736;          // 2,420,736
  float* Sb   = VT   + 2420736;          // 192*197*197         = 7,451,328 (7,449,888 reserved)
  float* AO   = Sb   + 7449888;          // 2,420,736
  float* MG   = AO   + 2420736;          // 2,420,736
  float* TMP  = MG   + 2420736;          // 3138*3072           = 9,639,936
  float* TMP2 = TMP  + 9639936;          // 2,420,736

  // ---- Patch embedding: im2row -> GEMM(conv) -> assemble X [B,1569,C] ----
  im2row_kernel<<<EL(BTN * CC)>>>(x, TMP, BTN * CC);
  run_gemm(stream, TMP, conv_w, conv_b, TMP2, BTN, CC, CC, 1, 0, 0, 0, 0);
  assemble_kernel<<<EL(ROWS * CC)>>>(TMP2, cls_token, pos_embed, time_embed, X, ROWS * CC);

  // --------------------------- 12 transformer blocks ----------------------
  for (int d = 0; d < 12; ++d) {
    const float* w_tn  = tn_w  + (size_t)d * CC;
    const float* b_tn  = tn_b  + (size_t)d * CC;
    const float* w_tq  = tqkv_w+ (size_t)d * 3 * CC * CC;
    const float* b_tq  = tqkv_b+ (size_t)d * 3 * CC;
    const float* w_tp  = tproj_w + (size_t)d * CC * CC;
    const float* b_tp  = tproj_b + (size_t)d * CC;
    const float* w_tf  = tfc_w   + (size_t)d * CC * CC;
    const float* b_tf  = tfc_b   + (size_t)d * CC;
    const float* w_n1  = n1_w + (size_t)d * CC;
    const float* b_n1  = n1_b + (size_t)d * CC;
    const float* w_sq  = sqkv_w + (size_t)d * 3 * CC * CC;
    const float* b_sq  = sqkv_b + (size_t)d * 3 * CC;
    const float* w_sp  = sproj_w + (size_t)d * CC * CC;
    const float* b_sp  = sproj_b + (size_t)d * CC;
    const float* w_n2  = n2_w + (size_t)d * CC;
    const float* b_n2  = n2_b + (size_t)d * CC;
    const float* w_f1  = fc1_w + (size_t)d * 4 * CC * CC;
    const float* b_f1  = fc1_b + (size_t)d * 4 * CC;
    const float* w_f2  = fc2_w + (size_t)d * CC * 4 * CC;
    const float* b_f2  = fc2_b + (size_t)d * CC;

    // ---- Temporal attention (per (b,n) sequence of length T=8) ----
    ln_kernel<<<ROWS, 256, 0, stream>>>(X, w_tn, b_tn, LNB, CC);
    // body rows only: per-batch A offset +C, stride SEQ*C
    run_gemm(stream, LNB + CC, w_tq, b_tq, QKV, TT * NP, 3 * CC, CC,
             BB, (long long)SEQ * CC, 0, (long long)TT * NP * 3 * CC, 0);
    split_t_kernel<<<EL(BTN * CC)>>>(QKV, Qb, Kb, VT, BTN * CC);
    run_gemm(stream, Qb, Kb, nullptr, Sb, LT, LT, HD,
             BHT, LT * HD, LT * HD, LT * LT, 0);
    softmax_kernel<<<BHT * LT, 256, 0, stream>>>(Sb, LT, 0.125f);
    run_gemm(stream, Sb, VT, nullptr, AO, LT, HD, LT,
             BHT, LT * LT, HD * LT, LT * HD, 0);
    merge_t_kernel<<<EL(BTN * CC)>>>(AO, MG, BTN * CC);
    run_gemm(stream, MG, w_tp, b_tp, TMP2, BTN, CC, CC, 1, 0, 0, 0, 0);
    run_gemm(stream, TMP2, w_tf, b_tf, MG, BTN, CC, CC, 1, 0, 0, 0, 0);
    add_body_kernel<<<EL(BTN * CC)>>>(X, MG, BTN * CC);

    // ---- Spatial attention (per (b,t) sequence of length 197) ----
    build_xs_kernel<<<EL(RS * CC)>>>(X, TMP, RS * CC);
    ln_kernel<<<RS, 256, 0, stream>>>(TMP, w_n1, b_n1, LNB, CC);
    run_gemm(stream, LNB, w_sq, b_sq, QKV, RS, 3 * CC, CC, 1, 0, 0, 0, 0);
    split_s_kernel<<<EL(RS * CC)>>>(QKV, Qb, Kb, VT, RS * CC);
    run_gemm(stream, Qb, Kb, nullptr, Sb, LS, LS, HD,
             BHS, LS * HD, LS * HD, LS * LS, 0);
    softmax_kernel<<<BHS * LS, 256, 0, stream>>>(Sb, LS, 0.125f);
    run_gemm(stream, Sb, VT, nullptr, AO, LS, HD, LS,
             BHS, LS * LS, HD * LS, LS * HD, 0);
    merge_s_kernel<<<EL(RS * CC)>>>(AO, MG, RS * CC);
    run_gemm(stream, MG, w_sp, b_sp, TMP2, RS, CC, CC, 1, 0, 0, 0, 0);
    combine_body_kernel<<<EL(BTN * CC)>>>(X, TMP2, BTN * CC);
    combine_cls_kernel<<<EL(BB * CC)>>>(X, TMP2, BB * CC);

    // ---- MLP (all B*SEQ tokens) ----
    ln_kernel<<<ROWS, 256, 0, stream>>>(X, w_n2, b_n2, LNB, CC);
    run_gemm(stream, LNB, w_f1, b_f1, TMP, ROWS, 4 * CC, CC, 1, 0, 0, 0, 1);  // +gelu
    run_gemm(stream, TMP, w_f2, b_f2, TMP2, ROWS, CC, 4 * CC, 1, 0, 0, 0, 0);
    add_all_kernel<<<EL(ROWS * CC)>>>(X, TMP2, ROWS * CC);
  }

  // ---- Final norm on cls tokens + classification head ----
  gather_cls_kernel<<<EL(BB * CC)>>>(X, LNB, BB * CC);
  ln_kernel<<<BB, 256, 0, stream>>>(LNB, norm_w, norm_b, QKV, CC);
  run_gemm(stream, QKV, head_w, head_b, (float*)d_out, BB, 1000, CC, 1, 0, 0, 0, 0);
}
